// BasicTransDecoderBlock_19404662243454
// MI455X (gfx1250) — compile-verified, matrix operand order checked by reading, not run on hardware
//
#include <hip/hip_runtime.h>

typedef __attribute__((ext_vector_type(16))) _Float16 v16h;
typedef __attribute__((ext_vector_type(8)))  _Float16 v8h;
typedef __attribute__((ext_vector_type(8)))  float    v8f;

#define BATCH   8
#define NHEADS  8
#define DHEAD   32

// Assemble one 16-bit WMMA A/B operand half for this lane.
// Operand layout (ISA 7.12.2): element e -> K = hi*8 + e, element e+8 -> K = 16 + hi*8 + e.
// p must already point at (row base + hi*8); both 8-half runs are 16B aligned.
__device__ __forceinline__ v16h ld16h_split(const _Float16* p) {
  v8h lo = *(const v8h*)(p);
  v8h hc = *(const v8h*)(p + 16);
  v16h r;
#pragma unroll
  for (int e = 0; e < 8; e++) { r[e] = lo[e]; r[e + 8] = hc[e]; }
  return r;
}

// ---------------- BatchNorm statistics: mean/rstd per channel over (B,H,W) ----
__global__ void bn_stats_kernel(const float* __restrict__ x, float* __restrict__ mean,
                                float* __restrict__ rstd, int C, int logHW, int n_total,
                                float eps) {
  __shared__ float ssum[256];
  __shared__ float ssq[256];
  int c = blockIdx.x;
  int HWm = (1 << logHW) - 1;
  float s = 0.f, q = 0.f;
  for (int i = threadIdx.x; i < n_total; i += 256) {
    int b = i >> logHW;
    int r = i & HWm;
    float v = x[(((b * C + c)) << logHW) + r];
    s += v;
    q += v * v;
  }
  ssum[threadIdx.x] = s;
  ssq[threadIdx.x]  = q;
  __syncthreads();
  for (int off = 128; off > 0; off >>= 1) {
    if (threadIdx.x < off) {
      ssum[threadIdx.x] += ssum[threadIdx.x + off];
      ssq[threadIdx.x]  += ssq[threadIdx.x + off];
    }
    __syncthreads();
  }
  if (threadIdx.x == 0) {
    float m = ssum[0] / (float)n_total;
    float v = ssq[0] / (float)n_total - m * m;
    mean[c] = m;
    rstd[c] = rsqrtf(v + eps);
  }
}

// ---------------- BatchNorm apply, f32 NCHW out -------------------------------
__global__ void bn_apply_kernel(const float* __restrict__ x, float* __restrict__ y,
                                const float* __restrict__ mean, const float* __restrict__ rstd,
                                const float* __restrict__ g, const float* __restrict__ bt,
                                int logC, int logHW, int total) {
  int idx = blockIdx.x * blockDim.x + threadIdx.x;
  if (idx >= total) return;
  int c = (idx >> logHW) & ((1 << logC) - 1);
  y[idx] = (x[idx] - mean[c]) * rstd[c] * g[c] + bt[c];
}

// ---------------- BatchNorm + ReLU, writes f16 transposed (pixel-major) -------
// yT[(b*HW + p)*C + c] = relu(bn(x[(b*C+c)*HW+p]))
__global__ void bn_relu_pack_kernel(const float* __restrict__ x, _Float16* __restrict__ yT,
                                    const float* __restrict__ mean, const float* __restrict__ rstd,
                                    const float* __restrict__ g, const float* __restrict__ bt,
                                    int logC, int logHW, int total) {
  int idx = blockIdx.x * blockDim.x + threadIdx.x;
  if (idx >= total) return;
  int p = idx & ((1 << logHW) - 1);
  int c = (idx >> logHW) & ((1 << logC) - 1);
  int b = idx >> (logHW + logC);
  float v = (x[idx] - mean[c]) * rstd[c] * g[c] + bt[c];
  v = fmaxf(v, 0.f);
  yT[(size_t)((((b << logHW) + p) << logC) + c)] = (_Float16)v;
}

// ---------------- pack activations NCHW f32 -> (b*HW+p, k) f16 ----------------
__global__ void pack_act_kernel(const float* __restrict__ x, _Float16* __restrict__ y,
                                int logK, int logHW, int total) {
  int idx = blockIdx.x * blockDim.x + threadIdx.x;
  if (idx >= total) return;
  int k = idx & ((1 << logK) - 1);
  int m = idx >> logK;
  int p = m & ((1 << logHW) - 1);
  int b = m >> logHW;
  y[idx] = (_Float16)x[(size_t)((((b << logK) + k) << logHW) + p)];
}

// ---------------- pack weights f32 -> f16 (same layout) -----------------------
__global__ void pack_w_kernel(const float* __restrict__ w, _Float16* __restrict__ y, int total) {
  int idx = blockIdx.x * blockDim.x + threadIdx.x;
  if (idx >= total) return;
  y[idx] = (_Float16)w[idx];
}

// ---------------- Depthwise 3x3 conv, pad 1; writes f16 transposed ------------
// yT[(b*HW + p)*C + c]
__global__ void dwconv3x3_kernel(const float* __restrict__ x, const float* __restrict__ w,
                                 _Float16* __restrict__ yT, int logC, int logH, int logW,
                                 int total) {
  int idx = blockIdx.x * blockDim.x + threadIdx.x;
  if (idx >= total) return;
  int W = 1 << logW, H = 1 << logH;
  int xw = idx & (W - 1);
  int t  = idx >> logW;
  int yy = t & (H - 1); t >>= logH;
  int c  = t & ((1 << logC) - 1);
  int b  = t >> logC;
  const float* xp = x + ((size_t)((b << logC) + c) << (logH + logW));
  const float* wp = w + c * 9;
  float acc = 0.f;
#pragma unroll
  for (int ky = 0; ky < 3; ky++) {
    int iy = yy + ky - 1;
    if (iy < 0 || iy >= H) continue;
#pragma unroll
    for (int kx = 0; kx < 3; kx++) {
      int ix = xw + kx - 1;
      if (ix < 0 || ix >= W) continue;
      acc += xp[(iy << logW) + ix] * wp[ky * 3 + kx];
    }
  }
  int p = (yy << logW) | xw;
  yT[(size_t)((((b << (logH + logW)) + p) << logC) + c)] = (_Float16)acc;
}

// ---------------- Bilinear resize (matches jnp.linspace endpoints) ------------
__global__ void resize_bilinear_kernel(const float* __restrict__ in, float* __restrict__ out,
                                       int Ctot, int coff, int logC, int Hc, int Wc,
                                       int logHo, int logWo, int total) {
  int idx = blockIdx.x * blockDim.x + threadIdx.x;
  if (idx >= total) return;
  int Wo = 1 << logWo, Ho = 1 << logHo;
  int ox = idx & (Wo - 1);
  int t  = idx >> logWo;
  int oy = t & (Ho - 1); t >>= logHo;
  int c  = t & ((1 << logC) - 1);
  int b  = t >> logC;
  float sy = (Ho > 1) ? (float)(Hc - 1) / (float)(Ho - 1) : 0.f;
  float sx = (Wo > 1) ? (float)(Wc - 1) / (float)(Wo - 1) : 0.f;
  float ys = oy * sy;
  float xs = ox * sx;
  int y0 = (int)floorf(ys); if (y0 > Hc - 1) y0 = Hc - 1;
  int x0 = (int)floorf(xs); if (x0 > Wc - 1) x0 = Wc - 1;
  int y1 = min(y0 + 1, Hc - 1);
  int x1 = min(x0 + 1, Wc - 1);
  float wy = ys - (float)y0;
  float wx = xs - (float)x0;
  const float* p = in + (size_t)((b * Ctot) + coff + c) * Hc * Wc;
  float r0 = p[y0 * Wc + x0] * (1.f - wx) + p[y0 * Wc + x1] * wx;
  float r1 = p[y1 * Wc + x0] * (1.f - wx) + p[y1 * Wc + x1] * wx;
  out[idx] = r0 * (1.f - wy) + r1 * wy;
}

// ---------------- split_heads (+ optional LN over DHEAD) -> f16 (b,h,i,d) -----
__global__ void ln_pack_kernel(const float* __restrict__ src, const float* __restrict__ g,
                               const float* __restrict__ bt, _Float16* __restrict__ dst,
                               int logHW, int total, int do_ln) {
  int idx = blockIdx.x * blockDim.x + threadIdx.x;
  if (idx >= total) return;
  int i = idx & ((1 << logHW) - 1);
  int h = (idx >> logHW) & (NHEADS - 1);
  int b = idx >> (logHW + 3);
  float vals[DHEAD];
  const float* sp = src + ((size_t)(b * 256) << logHW) + i;
#pragma unroll
  for (int d = 0; d < DHEAD; d++) vals[d] = sp[(size_t)(d * NHEADS + h) << logHW];
  _Float16 tmp[DHEAD];
  if (do_ln) {
    float m = 0.f;
#pragma unroll
    for (int d = 0; d < DHEAD; d++) m += vals[d];
    m *= (1.f / DHEAD);
    float v = 0.f;
#pragma unroll
    for (int d = 0; d < DHEAD; d++) { float dv = vals[d] - m; v += dv * dv; }
    v *= (1.f / DHEAD);
    float rs = rsqrtf(v + 1e-6f);
#pragma unroll
    for (int d = 0; d < DHEAD; d++)
      tmp[d] = (_Float16)((vals[d] - m) * rs * g[h * DHEAD + d] + bt[h * DHEAD + d]);
  } else {
#pragma unroll
    for (int d = 0; d < DHEAD; d++) tmp[d] = (_Float16)vals[d];
  }
  v8h* dp = (v8h*)(dst + (((size_t)(b * NHEADS + h) << logHW) + i) * DHEAD);
#pragma unroll
  for (int q8 = 0; q8 < 4; q8++) {
    v8h tv;
#pragma unroll
    for (int e = 0; e < 8; e++) tv[e] = tmp[q8 * 8 + e];
    dp[q8] = tv;
  }
}

// ---------------- pack V transposed: vT[(b,h,d,j)] f16 ------------------------
// src v16: (B,256,16,16) NCHW with channel = d*8+h
__global__ void pack_vT_kernel(const float* __restrict__ v16, _Float16* __restrict__ vT,
                               int total) {
  int idx = blockIdx.x * blockDim.x + threadIdx.x;
  if (idx >= total) return;
  int j = idx & 255;
  int h = (idx >> 8) & 7;
  int b = idx >> 11;
#pragma unroll
  for (int d = 0; d < DHEAD; d++)
    vT[(size_t)(((b * NHEADS + h) * DHEAD + d) << 8) + j] =
        (_Float16)v16[(size_t)((b * 256 + d * NHEADS + h) << 8) + j];
}

// ---------------- WMMA GEMM on packed f16 operands ----------------------------
// actT: (B*P, K) f16 row-major ; wT: (N, K) f16 row-major (B operand: col n, contig k)
// out (f32 NCHW): out[(b*N+n)*P + p] = sum_k actT[m,k]*wT[n,k] (+bias[n]) (+resid)
__global__ void gemm_wmma_kernel(const _Float16* __restrict__ actT, const _Float16* __restrict__ wT,
                                 const float* __restrict__ bias, const float* __restrict__ resid,
                                 float* __restrict__ out, int K, int N, int P, int logP) {
  int wave = threadIdx.x >> 5;
  int lane = threadIdx.x & 31;
  int Mt = (BATCH << logP) >> 4;
  int tile = blockIdx.x * 8 + wave;
  int mt = tile & (Mt - 1);
  int nt = tile >> (logP - 1);     // log2(Mt) = logP - 1
  int row = lane & 15;
  int hi  = lane >> 4;
  int hi8 = hi << 3;
  int m   = mt * 16 + row;
  int n   = nt * 16 + row;
  const _Float16* ap = actT + (size_t)m * K + hi8;
  const _Float16* wp = wT + (size_t)n * K + hi8;

  v8f acc = {};
  for (int k0 = 0; k0 < K; k0 += 32) {
    if (k0 + 32 < K) {
      __builtin_prefetch(ap + k0 + 32, 0, 0);
      __builtin_prefetch(wp + k0 + 32, 0, 0);
    }
    v16h a  = ld16h_split(ap + k0);
    v16h bm = ld16h_split(wp + k0);
    acc = __builtin_amdgcn_wmma_f32_16x16x32_f16(false, a, false, bm, (short)0, acc,
                                                 false, false);
  }
#pragma unroll
  for (int r = 0; r < 8; r++) {
    int mr = mt * 16 + r + 8 * hi;
    int br = mr >> logP;
    int pr = mr & ((1 << logP) - 1);
    float v = acc[r];
    if (bias)  v += bias[n];
    size_t oidx = (size_t)(((br * N + n) << logP) + pr);
    if (resid) v += resid[oidx];
    out[oidx] = v;
  }
}

// ---------------- Fused attention: o = ((q k^T + bias) / DHEAD) v -------------
// qh: (B,H,4096,32) f16 ; kh: (B,H,256,32) f16 ; vT: (B,H,32,256) f16
// o: (B,256,4096) f32 NCHW with c = d*8+h
__global__ void attention_kernel(const _Float16* __restrict__ qh, const _Float16* __restrict__ kh,
                                 const _Float16* __restrict__ vT, const float* __restrict__ rel,
                                 float* __restrict__ o) {
  __shared__ __align__(16) _Float16 smem[8][16][32];
  int wave = threadIdx.x >> 5;
  int lane = threadIdx.x & 31;
  int task = blockIdx.x * 8 + wave;
  int qt = task & 255;
  int h  = (task >> 8) & 7;
  int b  = task >> 11;
  int row = lane & 15;
  int hi  = lane >> 4;
  int hi8 = hi << 3;
  int col = lane & 15;
  int qbase = qt * 16;
  int bh = b * NHEADS + h;

  const _Float16* qp = qh + ((size_t)bh * 4096) * DHEAD;
  const _Float16* kp = kh + ((size_t)bh * 256) * DHEAD;
  const _Float16* vp = vT + ((size_t)bh * DHEAD) * 256;

  v16h aq = ld16h_split(qp + (qbase + row) * DHEAD + hi8);

  v8f acc0 = {}, acc1 = {};
  for (int jc = 0; jc < 8; jc++) {
    int j0 = jc * 32;
#pragma unroll
    for (int t = 0; t < 2; t++) {
      int jb = j0 + t * 16;
      v16h bk = ld16h_split(kp + (jb + col) * DHEAD + hi8);
      v8f z = {};
      v8f s = __builtin_amdgcn_wmma_f32_16x16x32_f16(false, aq, false, bk, (short)0, z,
                                                     false, false);
#pragma unroll
      for (int r = 0; r < 8; r++) {
        int i = qbase + r + 8 * hi;      // query pixel in 64x64
        int j = jb + col;                // kv index in 16x16
        int dy = ((i >> 6) >> 2) - (j >> 4) + 15;
        int dx = ((i & 63) >> 2) - (j & 15) + 15;
        float v = (s[r] + rel[(dy * 31 + dx) * NHEADS + h]) * (1.0f / DHEAD);
        smem[wave][r + 8 * hi][t * 16 + col] = (_Float16)v;
      }
    }
    asm volatile("" ::: "memory");
    v16h a2 = ld16h_split(&smem[wave][row][hi8]);
    asm volatile("" ::: "memory");
#pragma unroll
    for (int t = 0; t < 2; t++) {
      // B operand for A*V: column d = t*16+col, K index = j (contiguous in vT)
      v16h bv = ld16h_split(vp + ((t * 16 + col) << 8) + j0 + hi8);
      if (t == 0)
        acc0 = __builtin_amdgcn_wmma_f32_16x16x32_f16(false, a2, false, bv, (short)0,
                                                      acc0, false, false);
      else
        acc1 = __builtin_amdgcn_wmma_f32_16x16x32_f16(false, a2, false, bv, (short)0,
                                                      acc1, false, false);
    }
  }
#pragma unroll
  for (int r = 0; r < 8; r++) {
    int i = qbase + r + 8 * hi;
    o[(size_t)((b * 256 + col * 8 + h) << 12) + i]        = acc0[r];
    o[(size_t)((b * 256 + (16 + col) * 8 + h) << 12) + i] = acc1[r];
  }
}

// ---------------- workspace layout (float units) ------------------------------
#define OFF_STATS  0           // x1 m@0 rs@512, x2 m@1024 rs@1280, o2 m@1536 rs@1792
#define OFF_A      4096        // x2n -> o_buf                    (8,388,608)
#define OFF_B      8392704     // q -> o2 (res2)                  (8,388,608)
#define OFF_C      16781312    // qh f16 (8,388,608 halves)       (4,194,304 slots)
#define OFF_E      21*999936+0 // placeholder (recomputed below)
#undef  OFF_E
#define OFF_E      20975616    // residue                         (8,388,608)
#define OFF_F      29364224    // x1n -> kv                       (4,194,304)
#define OFF_H      33558528    // x1c -> {k16, v16, kh, vh}       (2,097,152)
#define OFF_K16    (OFF_H)
#define OFF_V16    (OFF_H + 524288)
#define OFF_KH     (OFF_H + 1048576)
#define OFF_VH     (OFF_H + 1310720)
#define OFF_PA     35655680    // packed activations f16          (4,194,304 slots)
#define OFF_PW     39849984    // packed weights f16              (131,072 slots)
// total = 39,981,056 floats ~= 160 MB

extern "C" void kernel_launch(void* const* d_in, const int* in_sizes, int n_in,
                              void* d_out, int out_size, void* d_ws, size_t ws_size,
                              hipStream_t stream) {
  const float* x1        = (const float*)d_in[0];
  const float* x2        = (const float*)d_in[1];
  const float* conv_ch_w = (const float*)d_in[2];
  const float* conv_ch_b = (const float*)d_in[3];
  const float* norm_l_g  = (const float*)d_in[4];
  const float* norm_l_b  = (const float*)d_in[5];
  const float* norm_h_g  = (const float*)d_in[6];
  const float* norm_h_b  = (const float*)d_in[7];
  const float* to_kv_dw  = (const float*)d_in[8];
  const float* to_kv_pw  = (const float*)d_in[9];
  const float* to_q_dw   = (const float*)d_in[10];
  const float* to_q_pw   = (const float*)d_in[11];
  const float* to_out_dw = (const float*)d_in[12];
  const float* to_out_pw = (const float*)d_in[13];
  const float* rel_table = (const float*)d_in[14];
  const float* normq_g   = (const float*)d_in[15];
  const float* normq_b   = (const float*)d_in[16];
  const float* normk_g   = (const float*)d_in[17];
  const float* normk_b   = (const float*)d_in[18];
  const float* norm2_g   = (const float*)d_in[19];
  const float* norm2_b   = (const float*)d_in[20];
  const float* mlp_w     = (const float*)d_in[21];
  float* out = (float*)d_out;
  float* ws  = (float*)d_ws;
  _Float16* pa = (_Float16*)(ws + OFF_PA);
  _Float16* pw = (_Float16*)(ws + OFF_PW);

  // 1. BN stats
  bn_stats_kernel<<<512, 256, 0, stream>>>(x1, ws + 0,    ws + 512,  512, 10, 8192,  1e-5f);
  bn_stats_kernel<<<256, 256, 0, stream>>>(x2, ws + 1024, ws + 1280, 256, 12, 32768, 1e-5f);

  // 2. x1c = conv1x1(x1) + b : pack x1 + w, GEMM (K=512,N=256,P=1024)
  pack_act_kernel<<<16384, 256, 0, stream>>>(x1, pa, 9, 10, 4194304);
  pack_w_kernel<<<512, 256, 0, stream>>>(conv_ch_w, pw, 131072);
  gemm_wmma_kernel<<<1024, 256, 0, stream>>>(pa, pw, conv_ch_b, nullptr,
                                             ws + OFF_H, 512, 256, 1024, 10);
  // 3. residue = bilinear(x1c, 64, 64)
  resize_bilinear_kernel<<<32768, 256, 0, stream>>>(ws + OFF_H, ws + OFF_E, 256, 0, 8,
                                                    32, 32, 6, 6, 8388608);
  // 4. x1n = bn(x1)
  bn_apply_kernel<<<16384, 256, 0, stream>>>(x1, ws + OFF_F, ws + 0, ws + 512,
                                             norm_l_g, norm_l_b, 9, 10, 4194304);
  // 5. kv = dsconv(x1n): depthwise -> packed f16, pointwise GEMM 512->512
  dwconv3x3_kernel<<<16384, 256, 0, stream>>>(ws + OFF_F, to_kv_dw, pa, 9, 5, 5, 4194304);
  pack_w_kernel<<<1024, 256, 0, stream>>>(to_kv_pw, pw, 262144);
  gemm_wmma_kernel<<<2048, 256, 0, stream>>>(pa, pw, nullptr, nullptr,
                                             ws + OFF_F, 512, 512, 1024, 10);
  // 6. x2n = bn(x2); q = dsconv(x2n) 256->256
  bn_apply_kernel<<<32768, 256, 0, stream>>>(x2, ws + OFF_A, ws + 1024, ws + 1280,
                                             norm_h_g, norm_h_b, 8, 12, 8388608);
  dwconv3x3_kernel<<<32768, 256, 0, stream>>>(ws + OFF_A, to_q_dw, pa, 8, 6, 6, 8388608);
  pack_w_kernel<<<256, 256, 0, stream>>>(to_q_pw, pw, 65536);
  gemm_wmma_kernel<<<4096, 256, 0, stream>>>(pa, pw, nullptr, nullptr,
                                             ws + OFF_B, 256, 256, 4096, 12);
  // 7. k16/v16 = bilinear(kv halves, 16x16)
  resize_bilinear_kernel<<<2048, 256, 0, stream>>>(ws + OFF_F, ws + OFF_K16, 512, 0,   8,
                                                   32, 32, 4, 4, 524288);
  resize_bilinear_kernel<<<2048, 256, 0, stream>>>(ws + OFF_F, ws + OFF_V16, 512, 256, 8,
                                                   32, 32, 4, 4, 524288);
  // 8. split heads + LN -> f16 packs; V transposed
  ln_pack_kernel<<<1024, 256, 0, stream>>>(ws + OFF_B, normq_g, normq_b,
                                           (_Float16*)(ws + OFF_C), 12, 262144, 1);
  ln_pack_kernel<<<64, 256, 0, stream>>>(ws + OFF_K16, normk_g, normk_b,
                                         (_Float16*)(ws + OFF_KH), 8, 16384, 1);
  pack_vT_kernel<<<64, 256, 0, stream>>>(ws + OFF_V16, (_Float16*)(ws + OFF_VH), 16384);
  // 9. fused attention -> o (NCHW, c = d*8+h)
  attention_kernel<<<2048, 256, 0, stream>>>((const _Float16*)(ws + OFF_C),
                                             (const _Float16*)(ws + OFF_KH),
                                             (const _Float16*)(ws + OFF_VH),
                                             rel_table, ws + OFF_A);
  // 10. o = dsconv(o) + residue
  dwconv3x3_kernel<<<32768, 256, 0, stream>>>(ws + OFF_A, to_out_dw, pa, 8, 6, 6, 8388608);
  pack_w_kernel<<<256, 256, 0, stream>>>(to_out_pw, pw, 65536);
  gemm_wmma_kernel<<<4096, 256, 0, stream>>>(pa, pw, nullptr, ws + OFF_E,
                                             ws + OFF_B, 256, 256, 4096, 12);
  // 11. o3 = relu(bn(o2)) -> packed f16; out = conv1x1(o3, mlp_w) + o2
  bn_stats_kernel<<<256, 256, 0, stream>>>(ws + OFF_B, ws + 1536, ws + 1792,
                                           256, 12, 32768, 1e-5f);
  bn_relu_pack_kernel<<<32768, 256, 0, stream>>>(ws + OFF_B, pa, ws + 1536, ws + 1792,
                                                 norm2_g, norm2_b, 8, 12, 8388608);
  pack_w_kernel<<<256, 256, 0, stream>>>(mlp_w, pw, 65536);
  gemm_wmma_kernel<<<4096, 256, 0, stream>>>(pa, pw, nullptr, ws + OFF_B,
                                             out, 256, 256, 4096, 12);
}
